// AttentionFreeTransformer_33878702031388
// MI455X (gfx1250) — matrix operand order repaired and checked by reading, compiled, not run on hardware
//
#include <hip/hip_runtime.h>
#include <hip/hip_bf16.h>

// ---------------------------------------------------------------------------
// AFT forward for MI455X (gfx1250, wave32, WMMA).
// exp_pos_bias == 1  =>  num/den are per-(b,d) reductions over t.
// out[b,i,:] = (sigmoid(q[b,i,:]@Wq) * num[b,:]/den[b,:]) @ Wo
//
// f32->bf16 conversion hoisted to one-time prep passes; GEMM K-loops are
// software-pipelined (double-buffered fragments) so loads for step k+1
// overlap the v_wmma ops of step k.
// ---------------------------------------------------------------------------

#define B_  4
#define T_  2048
#define DM_ 1024
#define DH_ 256

typedef __attribute__((ext_vector_type(16))) __bf16 v16bf;
typedef __attribute__((ext_vector_type(8)))  float  v8f;

union FragU { v16bf v; uint4 q[2]; };

__device__ __forceinline__ unsigned short f2bfu(float f) {
    unsigned u = __builtin_bit_cast(unsigned, f);
    u += 0x7FFFu + ((u >> 16) & 1u);            // round-to-nearest-even
    return (unsigned short)(u >> 16);
}
__device__ __forceinline__ __bf16 f2bf(float f) {
    unsigned short s = f2bfu(f);
    return __builtin_bit_cast(__bf16, s);
}

#define WMMA_BF16(A, Bf, C) \
    __builtin_amdgcn_wmma_f32_16x16x32_bf16(false, (A), false, (Bf), (short)0, (C), false, false)

// A fragment (16x32 bf16) from bf16 row-major [lda]: two 16B loads per lane.
__device__ __forceinline__ v16bf load_a_bf16(const __bf16* __restrict__ A, int lda,
                                             int m0, int k0, int lane) {
    const int hl = lane >> 4, m = lane & 15;
    const __bf16* ap = A + (size_t)(m0 + m) * lda + k0 + hl * 8;
    FragU u;
    u.q[0] = *(const uint4*)(ap);
    u.q[1] = *(const uint4*)(ap + 16);
    return u.v;
}

// Same fragment from bf16 (u16) in LDS: two ds_load_b128 per lane.
__device__ __forceinline__ v16bf load_a_lds(const unsigned short* A, int lda,
                                            int k0, int lane) {
    const int hl = lane >> 4, m = lane & 15;
    const unsigned short* ap = A + (size_t)m * lda + k0 + hl * 8;
    FragU u;
    u.q[0] = *(const uint4*)(ap);
    u.q[1] = *(const uint4*)(ap + 16);
    return u.v;
}

// B fragment (32x16 bf16) from pre-packed layout [kt][nt][lane][16].
__device__ __forceinline__ v16bf load_b_pack(const __bf16* __restrict__ Wp, int ntiles,
                                             int kt, int nt, int lane) {
    const __bf16* bp = Wp + (((size_t)kt * ntiles + nt) * 32 + lane) * 16;
    FragU u;
    u.q[0] = *(const uint4*)(bp);
    u.q[1] = *(const uint4*)(bp + 8);
    return u.v;
}

// ---------------------------------------------------------------------------
// Prep 1: elementwise f32 -> bf16 (8 elements/thread, b128 in / b128 out).
// ---------------------------------------------------------------------------
__global__ __launch_bounds__(256)
void cvt_bf16(const float* __restrict__ src, __bf16* __restrict__ dst, int n) {
    const int i = (blockIdx.x * 256 + threadIdx.x) * 8;
    if (i >= n) return;
    const float4 a = *(const float4*)(src + i);
    const float4 b = *(const float4*)(src + i + 4);
    uint4 o;
    o.x = (unsigned)f2bfu(a.x) | ((unsigned)f2bfu(a.y) << 16);
    o.y = (unsigned)f2bfu(a.z) | ((unsigned)f2bfu(a.w) << 16);
    o.z = (unsigned)f2bfu(b.x) | ((unsigned)f2bfu(b.y) << 16);
    o.w = (unsigned)f2bfu(b.z) | ((unsigned)f2bfu(b.w) << 16);
    *(uint4*)(dst + i) = o;
}

// ---------------------------------------------------------------------------
// Prep 2: pack W[Kd,N] f32 into B-fragment layout [Kd/32][N/16][32][16] bf16.
// ---------------------------------------------------------------------------
__global__ __launch_bounds__(256)
void pack_w(const float* __restrict__ W, __bf16* __restrict__ Wp, int Kd, int N) {
    const int ntiles = N >> 4;
    const int total = (Kd >> 5) * ntiles * 32;
    const int idx = blockIdx.x * 256 + threadIdx.x;
    if (idx >= total) return;
    const int lane = idx & 31;
    const int t    = idx >> 5;
    const int nt   = t % ntiles;
    const int kt   = t / ntiles;
    const int hl = lane >> 4, n = lane & 15;
    __bf16* dst = Wp + (size_t)idx * 16;
    const float* src = W + (size_t)(kt * 32 + hl * 8) * N + nt * 16 + n;
#pragma unroll
    for (int j = 0; j < 16; ++j) dst[j] = f2bf(src[(size_t)((j & 7) + (j >> 3) * 16) * N]);
}

// ---------------------------------------------------------------------------
// Kernel 1: C[M,N] = A[M,Kd] @ W[Kd,N]  (bf16 in, f32 out).
// Wave: 64x16 strip; K-loop double-buffered: next loads overlap current wmma.
// ---------------------------------------------------------------------------
__global__ __launch_bounds__(256)
void gemm_bf(const __bf16* __restrict__ A, const __bf16* __restrict__ Wp,
             float* __restrict__ C, int M, int Kd, int N) {
    const int lane = threadIdx.x & 31;
    const int wave = threadIdx.x >> 5;
    const int ntiles = N >> 4;
    const int gw = blockIdx.x * 8 + wave;
    const int m0 = (gw / ntiles) * 64;
    const int nt = gw % ntiles;
    const int n0 = nt * 16;
    if (m0 >= M) return;

    const int ktiles = Kd >> 5;

    // prologue: k-step 0 in flight
    v16bf bw = load_b_pack(Wp, ntiles, 0, nt, lane);
    v16bf a0 = load_a_bf16(A, Kd, m0,      0, lane);
    v16bf a1 = load_a_bf16(A, Kd, m0 + 16, 0, lane);
    v16bf a2 = load_a_bf16(A, Kd, m0 + 32, 0, lane);
    v16bf a3 = load_a_bf16(A, Kd, m0 + 48, 0, lane);

    v8f acc0 = {}, acc1 = {}, acc2 = {}, acc3 = {};
    for (int kt = 0; kt < ktiles - 1; ++kt) {
        const int k1 = (kt + 1) * 32;
        // issue next k-step's loads first ...
        v16bf nb  = load_b_pack(Wp, ntiles, kt + 1, nt, lane);
        v16bf na0 = load_a_bf16(A, Kd, m0,      k1, lane);
        v16bf na1 = load_a_bf16(A, Kd, m0 + 16, k1, lane);
        v16bf na2 = load_a_bf16(A, Kd, m0 + 32, k1, lane);
        v16bf na3 = load_a_bf16(A, Kd, m0 + 48, k1, lane);
        // ... then compute on current (already-arrived) fragments
        acc0 = WMMA_BF16(a0, bw, acc0);
        acc1 = WMMA_BF16(a1, bw, acc1);
        acc2 = WMMA_BF16(a2, bw, acc2);
        acc3 = WMMA_BF16(a3, bw, acc3);
        bw = nb; a0 = na0; a1 = na1; a2 = na2; a3 = na3;
    }
    // epilogue
    acc0 = WMMA_BF16(a0, bw, acc0);
    acc1 = WMMA_BF16(a1, bw, acc1);
    acc2 = WMMA_BF16(a2, bw, acc2);
    acc3 = WMMA_BF16(a3, bw, acc3);

    const int n = lane & 15, hl = lane >> 4;
#pragma unroll
    for (int r = 0; r < 8; ++r) {
        const int m = r + 8 * hl;
        C[(size_t)(m0 +      m) * N + n0 + n] = acc0[r];
        C[(size_t)(m0 + 16 + m) * N + n0 + n] = acc1[r];
        C[(size_t)(m0 + 32 + m) * N + n0 + n] = acc2[r];
        C[(size_t)(m0 + 48 + m) * N + n0 + n] = acc3[r];
    }
}

// ---------------------------------------------------------------------------
// Kernel 2: num[b,d] = sum_t exp(K[b,t,d]-max_b K)*V ; den likewise.
// ---------------------------------------------------------------------------
__global__ __launch_bounds__(256)
void reduce_kv(const float* __restrict__ Kp, const float* __restrict__ Vp,
               float* __restrict__ num, float* __restrict__ den) {
    const int d  = threadIdx.x;
    const int t0 = blockIdx.x * 64;
    float n0 = 0.f, n1 = 0.f, n2 = 0.f, n3 = 0.f;
    float d0 = 0.f, d1 = 0.f, d2 = 0.f, d3 = 0.f;
    for (int t = t0; t < t0 + 64; ++t) {
        const size_t o = (size_t)t * DH_ + d;
        const float k0 = Kp[o];
        const float k1 = Kp[o + (size_t)1 * T_ * DH_];
        const float k2 = Kp[o + (size_t)2 * T_ * DH_];
        const float k3 = Kp[o + (size_t)3 * T_ * DH_];
        const float mx = fmaxf(fmaxf(k0, k1), fmaxf(k2, k3));
        const float e0 = __expf(k0 - mx);
        const float e1 = __expf(k1 - mx);
        const float e2 = __expf(k2 - mx);
        const float e3 = __expf(k3 - mx);
        n0 += e0 * Vp[o];
        n1 += e1 * Vp[o + (size_t)1 * T_ * DH_];
        n2 += e2 * Vp[o + (size_t)2 * T_ * DH_];
        n3 += e3 * Vp[o + (size_t)3 * T_ * DH_];
        d0 += e0; d1 += e1; d2 += e2; d3 += e3;
    }
    atomicAdd(&num[0 * DH_ + d], n0);
    atomicAdd(&num[1 * DH_ + d], n1);
    atomicAdd(&num[2 * DH_ + d], n2);
    atomicAdd(&num[3 * DH_ + d], n3);
    atomicAdd(&den[0 * DH_ + d], d0);
    atomicAdd(&den[1 * DH_ + d], d1);
    atomicAdd(&den[2 * DH_ + d], d2);
    atomicAdd(&den[3 * DH_ + d], d3);
}

// ---------------------------------------------------------------------------
// Kernel 3: per 16-row tile: Q=qWq (WMMA), y=sigmoid(Q)*r -> LDS (bf16),
// then out = y @ Wo (WMMA, A fragments via ds_load_b128). Both K-loops
// double-buffered.
// ---------------------------------------------------------------------------
__global__ __launch_bounds__(256)
void fused_out(const __bf16* __restrict__ qbf, const __bf16* __restrict__ Wqp,
               const __bf16* __restrict__ Wop, const float* __restrict__ num,
               const float* __restrict__ den, float* __restrict__ out) {
    __shared__ unsigned short sY[16 * DH_];   // 8 KB: sigmoid(Q)*r as bf16
    __shared__ float          sr[DH_];        // 1 KB

    const int tid  = threadIdx.x;
    const int lane = tid & 31;
    const int wave = tid >> 5;
    const int row0 = blockIdx.x * 16;
    const int b    = row0 / T_;                 // tile never straddles a batch

    sr[tid] = num[b * DH_ + tid] / den[b * DH_ + tid];
    __syncthreads();

    // Phase 1: Q tile 16x256; wave handles two 16-wide column tiles.
    {
        const int hl = lane >> 4, n = lane & 15;
#pragma unroll
        for (int j = 0; j < 2; ++j) {
            const int nt = wave * 2 + j;
            const int n0 = nt * 16;
            v16bf a  = load_a_bf16(qbf, DM_, row0, 0, lane);
            v16bf bw = load_b_pack(Wqp, DH_ >> 4, 0, nt, lane);
            v8f acc = {};
            for (int kt = 0; kt < (DM_ >> 5) - 1; ++kt) {
                v16bf na = load_a_bf16(qbf, DM_, row0, (kt + 1) * 32, lane);
                v16bf nb = load_b_pack(Wqp, DH_ >> 4, kt + 1, nt, lane);
                acc = WMMA_BF16(a, bw, acc);
                a = na; bw = nb;
            }
            acc = WMMA_BF16(a, bw, acc);
#pragma unroll
            for (int r = 0; r < 8; ++r) {
                const int m = r + 8 * hl;
                const float sg = 1.0f / (1.0f + __expf(-acc[r]));
                sY[m * DH_ + n0 + n] = f2bfu(sg * sr[n0 + n]);
            }
        }
    }
    __syncthreads();

    // Phase 2: out tile 16x1024; wave handles eight 16-wide column tiles.
    for (int j = 0; j < 8; ++j) {
        const int nt = wave * 8 + j;
        const int n0 = nt * 16;
        v16bf a  = load_a_lds(sY, DH_, 0, lane);
        v16bf bw = load_b_pack(Wop, DM_ >> 4, 0, nt, lane);
        v8f acc = {};
#pragma unroll
        for (int kt = 0; kt < (DH_ >> 5) - 1; ++kt) {
            v16bf na = load_a_lds(sY, DH_, (kt + 1) * 32, lane);
            v16bf nb = load_b_pack(Wop, DM_ >> 4, kt + 1, nt, lane);
            acc = WMMA_BF16(a, bw, acc);
            a = na; bw = nb;
        }
        acc = WMMA_BF16(a, bw, acc);
        const int n = lane & 15, hl = lane >> 4;
#pragma unroll
        for (int r = 0; r < 8; ++r)
            out[((size_t)row0 + r + 8 * hl) * DM_ + n0 + n] = acc[r];
    }
}

// ---------------------------------------------------------------------------
extern "C" void kernel_launch(void* const* d_in, const int* in_sizes, int n_in,
                              void* d_out, int out_size, void* d_ws, size_t ws_size,
                              hipStream_t stream) {
    (void)in_sizes; (void)n_in; (void)out_size; (void)ws_size;
    const float* q  = (const float*)d_in[0];
    const float* k  = (const float*)d_in[1];
    const float* v  = (const float*)d_in[2];
    const float* Wq = (const float*)d_in[3];
    const float* Wk = (const float*)d_in[4];
    const float* Wv = (const float*)d_in[5];
    const float* Wo = (const float*)d_in[6];
    // d_in[7] = W_bias: exp(bias - max over a size-1 axis) == 1, so it cancels.

    const size_t act = (size_t)B_ * T_ * DM_;        // 8,388,608 elements
    const size_t kvw = (size_t)B_ * T_ * DH_;        // 2,097,152 elements

    char* ws = (char*)d_ws;
    __bf16* qbf  = (__bf16*)ws;            ws += act * 2;          // 16 MB
    __bf16* kbf  = (__bf16*)ws;            ws += act * 2;          // 16 MB
    __bf16* vbf  = (__bf16*)ws;            ws += act * 2;          // 16 MB
    float*  Kp   = (float*)ws;             ws += kvw * 4;          //  8 MB
    float*  Vp   = (float*)ws;             ws += kvw * 4;          //  8 MB
    __bf16* Wqp  = (__bf16*)ws;            ws += (size_t)DM_ * DH_ * 2;
    __bf16* Wkp  = (__bf16*)ws;            ws += (size_t)DM_ * DH_ * 2;
    __bf16* Wvp  = (__bf16*)ws;            ws += (size_t)DM_ * DH_ * 2;
    __bf16* Wop  = (__bf16*)ws;            ws += (size_t)DH_ * DM_ * 2;
    float*  num  = (float*)ws;             ws += (size_t)B_ * DH_ * 4;
    float*  den  = (float*)ws;
    float*  o    = (float*)d_out;

    const int M = B_ * T_;
    const int cvt_blocks  = (int)(act / 8 / 256);                  // 4096
    const int pack_blocks = ((DM_ / 32) * (DH_ / 16) * 32 + 255) / 256;
    const int gemm_blocks = (M / 64) * (DH_ / 16) / 8;             // 256

    cvt_bf16<<<cvt_blocks, 256, 0, stream>>>(q, qbf, (int)act);
    cvt_bf16<<<cvt_blocks, 256, 0, stream>>>(k, kbf, (int)act);
    cvt_bf16<<<cvt_blocks, 256, 0, stream>>>(v, vbf, (int)act);
    pack_w<<<pack_blocks, 256, 0, stream>>>(Wq, Wqp, DM_, DH_);
    pack_w<<<pack_blocks, 256, 0, stream>>>(Wk, Wkp, DM_, DH_);
    pack_w<<<pack_blocks, 256, 0, stream>>>(Wv, Wvp, DM_, DH_);
    pack_w<<<pack_blocks, 256, 0, stream>>>(Wo, Wop, DH_, DM_);

    gemm_bf<<<gemm_blocks, 256, 0, stream>>>(kbf, Wkp, Kp, M, DM_, DH_);
    gemm_bf<<<gemm_blocks, 256, 0, stream>>>(vbf, Wvp, Vp, M, DM_, DH_);

    hipMemsetAsync(num, 0, (size_t)2 * B_ * DH_ * sizeof(float), stream);
    reduce_kv<<<T_ / 64, 256, 0, stream>>>(Kp, Vp, num, den);
    fused_out<<<M / 16, 256, 0, stream>>>(qbf, Wqp, Wop, num, den, o);
}
